// Generator_47751446397482
// MI455X (gfx1250) — matrix operand order repaired and checked by reading, not compiled, not run on hardware
//
#include <hip/hip_runtime.h>
#include <hip/hip_bf16.h>

// SDE-GAN generator for MI455X (gfx1250, wave32, WMMA).
// Each workgroup owns a 16-row batch tile for the entire 127-step
// reversible-Heun scan (carry is independent per batch row). All MLP weights
// are converted to bf16 and pre-swizzled into WMMA B-fragment layout in LDS
// once (320 KB LDS/WGP fits gw2+gw1+gw0+activations exactly), so the
// sequential steps never touch global memory for weights. The (B,64,16)
// diffusion output never leaves WMMA accumulators: both matvecs g.dW_n and
// g.dW_{n+1} are fused into the layer-3 epilogue via lane shuffles.
// This round: hardware v_tanh_f32 (CDNA5 TRANS op), packed bf16 conversion,
// and raw v_sqrt_f32 to shrink the serial epilogue further.

#define BATCH 4096
#define T     128
#define HID   64
#define NZ    16
#define MLPW  128

typedef __attribute__((ext_vector_type(16))) __bf16 v16bf;
typedef __attribute__((ext_vector_type(8)))  float  v8f;

__device__ __forceinline__ float fast_rcp(float x) {
#if __has_builtin(__builtin_amdgcn_rcpf)
  return __builtin_amdgcn_rcpf(x);             // v_rcp_f32
#else
  return 1.0f / x;
#endif
}

__device__ __forceinline__ float fast_sqrt(float x) {
#if __has_builtin(__builtin_amdgcn_sqrtf)
  return __builtin_amdgcn_sqrtf(x);            // v_sqrt_f32
#else
  return sqrtf(x);
#endif
}

__device__ __forceinline__ float fast_tanh(float x) {
#if __has_builtin(__builtin_amdgcn_tanhf)
  return __builtin_amdgcn_tanhf(x);            // v_tanh_f32
#else
  // CDNA5 has a TANH transcendental (ISA ch.5 TRANS list); inline asm since
  // no builtin is exposed. Trailing v_nop covers the TRANS result hazard
  // (1 independent op required before the destination may be consumed).
  float r;
  asm volatile("v_tanh_f32 %0, %1\n\tv_nop" : "=v"(r) : "v"(x));
  return r;
#endif
}

__device__ __forceinline__ unsigned short f2bfu(float f) {
  unsigned u = __float_as_uint(f);
  u += 0x7FFFu + ((u >> 16) & 1u);              // round-to-nearest-even
  return (unsigned short)(u >> 16);
}

// Pack two floats into one dword of bf16 (low = a, high = b).
__device__ __forceinline__ unsigned pack2bf(float a, float b) {
#if __has_builtin(__builtin_amdgcn_cvt_pk_bf16_f32)
  typedef __attribute__((ext_vector_type(2))) __bf16 v2bf;
  v2bf p = __builtin_amdgcn_cvt_pk_bf16_f32(a, b);
  return __builtin_bit_cast(unsigned, p);       // v_cvt_pk_bf16_f32
#else
  return (unsigned)f2bfu(a) | ((unsigned)f2bfu(b) << 16);
#endif
}

__device__ __forceinline__ float lipswish(float x) {
  // 0.909 * x * sigmoid(x); sigmoid via v_exp + v_rcp
  return 0.909f * x * fast_rcp(1.0f + __expf(-x));
}

__device__ __forceinline__ v16bf ld_frag(const unsigned short* p) {
  return *(const v16bf*)p;                      // 32B -> ds_load_b128 x2
}

__device__ __forceinline__ v8f wmma_bf16(v16bf a, v16bf b, v8f c) {
  // D = A(16x32) * B(32x16) + C, f32 accumulate
  return __builtin_amdgcn_wmma_f32_16x16x32_bf16(false, a, false, b,
                                                 (short)0, c, false, false);
}

// A-fragment element placement for 16-bit 16x32 A tiles (ISA 7.12.2):
// lanes 0-15 hold K in {0..7,16..23}, lanes 16-31 hold K in {8..15,24..31}.
__device__ __forceinline__ void scatterA(unsigned short* A, int b, int j,
                                         unsigned short v) {
  int kk = j >> 5, kp = j & 31;
  int ln = b + (((kp >> 3) & 1) << 4);
  int ii = (kp & 7) + ((kp >= 16) ? 8 : 0);
  A[(kk * 32 + ln) * 16 + ii] = v;
}

// Pre-swizzle a KxN f32 row-major weight matrix into bf16 B-fragments:
// slot layout [(nt*KB+kb)*32 + lane]*16 + i, element (k = kb*32 + (i<8 ? i : i+8) + 8*(lane>=16),
// n = nt*16 + (lane&15)) so each lane's 16 values are 32B contiguous.
__device__ __forceinline__ void fill_b_frags(unsigned short* dst,
                                             const float* __restrict__ W,
                                             int K, int N, int tid) {
  int KB = K >> 5;
  int total = (N >> 4) * KB * 512;
  for (int s = tid; s < total; s += 256) {
    int i   = s & 15;
    int ln  = (s >> 4) & 31;
    int blk = s >> 9;
    int kb  = blk % KB;
    int nt  = blk / KB;
    int k = kb * 32 + (i < 8 ? i : i + 8) + ((ln >= 16) ? 8 : 0);
    int n = nt * 16 + (ln & 15);
    dst[s] = f2bfu(W[(size_t)k * N + n]);
  }
}

// Pack a thread's 4 consecutive zhat values (fixed b, h0..h0+3) into the
// layer-1 A-fragment region (one 8-byte ds_store).
__device__ __forceinline__ void write_tx(unsigned short* TX, int bown, int hown,
                                         const float* zh) {
  int kk = hown >> 5, kp = hown & 31;
  int ln = bown + (((kp >> 3) & 1) << 4);
  int i0 = (kp & 7) + ((kp >= 16) ? 8 : 0);
  uint2 pk;
  pk.x = pack2bf(zh[0], zh[1]);
  pk.y = pack2bf(zh[2], zh[3]);
  *(uint2*)&TX[(kk * 32 + ln) * 16 + i0] = pk;
}

__device__ __forceinline__ void readout(float* __restrict__ out, int bg, int t,
                                        const float* z, const float* rwreg,
                                        float rbv, int tid) {
  float s = z[0]*rwreg[0] + z[1]*rwreg[1] + z[2]*rwreg[2] + z[3]*rwreg[3];
  s += __shfl_xor(s, 1);
  s += __shfl_xor(s, 2);
  s += __shfl_xor(s, 4);
  s += __shfl_xor(s, 8);
  if ((tid & 15) == 0) out[(size_t)bg * (T * 2) + t * 2 + 1] = s + rbv;
}

// ---------------------------------------------------------------------------
// Kernel 1: initial MLP x0 = MLP(init_noise): 64 -> 128 -> 128 -> 64 (no tanh)
// ---------------------------------------------------------------------------
__global__ void sdegan_init_mlp(const float* __restrict__ x_in,
                                const float* __restrict__ w0, const float* __restrict__ b0,
                                const float* __restrict__ w1, const float* __restrict__ b1,
                                const float* __restrict__ w2, const float* __restrict__ b2,
                                float* __restrict__ x0_out) {
  extern __shared__ char smem[];
  unsigned short* W0 = (unsigned short*)smem;   // 8 nt * 2 kb * 512 = 8192 us
  unsigned short* W1 = W0 + 8192;               // 8 nt * 4 kb * 512 = 16384 us
  unsigned short* W2 = W1 + 16384;              // 4 nt * 4 kb * 512 = 8192 us
  unsigned short* XF = W2 + 8192;               // A-frags K=64: 1024 us
  unsigned short* H1 = XF + 1024;               // A-frags K=128: 2048 us
  unsigned short* H2 = H1 + 2048;               // A-frags K=128: 2048 us

  int tid = threadIdx.x;
  int wv = tid >> 5, lane = tid & 31;
  int kcol = lane & 15, hi = lane >> 4;
  int bbase = blockIdx.x * 16;

  fill_b_frags(W0, w0, 64, 128, tid);
  fill_b_frags(W1, w1, 128, 128, tid);
  fill_b_frags(W2, w2, 128, 64, tid);
  for (int s = tid; s < 1024; s += 256) {       // stage x as A-fragments
    int i = s & 15, ln = (s >> 4) & 31, kk = s >> 9;
    int k = kk * 32 + (i < 8 ? i : i + 8) + ((ln >= 16) ? 8 : 0);
    int m = ln & 15;
    XF[s] = f2bfu(x_in[(size_t)(bbase + m) * 64 + k]);
  }
  float b0j = b0[wv * 16 + kcol];
  float b1j = b1[wv * 16 + kcol];
  float b2j = (wv < 4) ? b2[wv * 16 + kcol] : 0.f;
  __syncthreads();

  { // layer 1 (wave w owns N-tile w)
    v8f acc = {};
    for (int kk = 0; kk < 2; ++kk)
      acc = wmma_bf16(ld_frag(XF + (kk * 32 + lane) * 16),
                      ld_frag(W0 + ((wv * 2 + kk) * 32 + lane) * 16), acc);
    for (int r = 0; r < 8; ++r)
      scatterA(H1, r + 8 * hi, wv * 16 + kcol, f2bfu(lipswish(acc[r] + b0j)));
  }
  __syncthreads();
  { // layer 2
    v8f acc = {};
    for (int kb = 0; kb < 4; ++kb)
      acc = wmma_bf16(ld_frag(H1 + (kb * 32 + lane) * 16),
                      ld_frag(W1 + ((wv * 4 + kb) * 32 + lane) * 16), acc);
    for (int r = 0; r < 8; ++r)
      scatterA(H2, r + 8 * hi, wv * 16 + kcol, f2bfu(lipswish(acc[r] + b1j)));
  }
  __syncthreads();
  if (wv < 4) { // layer 3, N = 64 -> 4 N-tiles
    v8f acc = {};
    for (int kb = 0; kb < 4; ++kb)
      acc = wmma_bf16(ld_frag(H2 + (kb * 32 + lane) * 16),
                      ld_frag(W2 + ((wv * 4 + kb) * 32 + lane) * 16), acc);
    for (int r = 0; r < 8; ++r)
      x0_out[(size_t)(bbase + r + 8 * hi) * 64 + wv * 16 + kcol] = acc[r] + b2j;
  }
}

// ---------------------------------------------------------------------------
// Kernel 2: full reversible-Heun scan. 16 batch rows/WG, 256 threads (8 waves),
// 320KB dynamic LDS. Weights LDS-resident for all 127 steps.
// ---------------------------------------------------------------------------
__global__ void sdegan_scan(const float* __restrict__ ts,
                            const float* __restrict__ dW,
                            const float* __restrict__ gw0, const float* __restrict__ gb0,
                            const float* __restrict__ gw1, const float* __restrict__ gb1,
                            const float* __restrict__ gw2, const float* __restrict__ gb2,
                            const float* __restrict__ rw,  const float* __restrict__ rb,
                            const float* __restrict__ x0,
                            float* __restrict__ out) {
  extern __shared__ char smem[];
  unsigned short* W2 = (unsigned short*)smem;   // 64 nt * 4 kb * 512 = 131072 us (256KB)
  unsigned short* W1 = W2 + 131072;             // 16384 us (32KB)
  unsigned short* W0 = W1 + 16384;              // 8192 us (16KB) : gw0 rows 1..64
  unsigned short* H1 = W0 + 8192;               // 2048 us (4KB)
  unsigned short* H2 = H1 + 2048;               // 2048 us (4KB); TX aliases [0..1023]
  float* BUF0 = (float*)(H2 + 2048);            // 1024 f32: a_prev / p_next
  float* BUF1 = BUF0 + 1024;                    // 1024 f32: p_cur
  // total = 262144+32768+16384+4096+4096+4096+4096 = 327680 B = 320 KB exactly

  int tid = threadIdx.x;
  int wv = tid >> 5, lane = tid & 31;
  int kcol = lane & 15, hi = lane >> 4;
  int bbase = blockIdx.x * 16;
  int bown = tid >> 4;                          // this thread's batch row (0..15)
  int hown = (tid & 15) * 4;                    // this thread's 4 hidden dims

  fill_b_frags(W2, gw2, 128, 1024, tid);
  fill_b_frags(W1, gw1, 128, 128, tid);
  fill_b_frags(W0, gw0 + 128, 64, 128, tid);    // skip t-row of gw0

  for (int i = 0; i < 8; ++i) {                 // ts column of output (B,T,2)
    int idx = tid + 256 * i;
    int t = idx & 127, bl = idx >> 7;
    out[(size_t)(bbase + bl) * (T * 2) + t * 2] = ts[t];
  }

  // hoisted per-lane constants
  float gb0j  = gb0[wv * 16 + kcol];
  float gw00j = gw0[wv * 16 + kcol];            // t-row of gw0 (rank-1 bias)
  float gb1j  = gb1[wv * 16 + kcol];
  float gb2r[8];
  for (int ti = 0; ti < 8; ++ti) gb2r[ti] = gb2[(wv + ti * 8) * 16 + kcol];
  float rwreg[4];
  for (int i = 0; i < 4; ++i) rwreg[i] = rw[hown + i];
  float rbv = rb[0];

  // carry state in registers: z, zhat (4 elements per thread)
  float z[4], zh[4];
  {
    const float4 zv = *(const float4*)&x0[(size_t)blockIdx.x * 1024 + tid * 4];
    z[0] = zh[0] = zv.x; z[1] = zh[1] = zv.y;
    z[2] = zh[2] = zv.z; z[3] = zh[3] = zv.w;
  }
  write_tx(H2, bown, hown, zh);                 // layer-1 A input = x0
  readout(out, bbase + bown, 0, z, rwreg, rbv, tid);

  // Iteration n evaluates g at t_{n+1}; n == -1 is the g0 prologue.
  for (int n = -1; n <= T - 2; ++n) {
    if (n >= 0) {
      const float4 ap = *(const float4*)&BUF0[tid * 4];  // a = g_n . dWs_n
      float a4[4] = {ap.x, ap.y, ap.z, ap.w};
      for (int i = 0; i < 4; ++i) {
        float zh_new = 2.f * z[i] - zh[i] + a4[i];
        z[i] += 0.5f * a4[i];
        zh[i] = zh_new;
      }
      write_tx(H2, bown, hown, zh);
    }
    __syncthreads();

    bool hasCur  = (n >= 0);
    bool hasNext = (n <= T - 3);
    float dwn[8] = {}, dwm[8] = {};
    if (hasCur) {
      float sc = fast_sqrt(ts[n + 1] - ts[n]);
      for (int r = 0; r < 8; ++r)
        dwn[r] = dW[((size_t)n * BATCH + bbase + r + 8 * hi) * NZ + kcol] * sc;
    }
    if (hasNext) {
      float sc = fast_sqrt(ts[n + 2] - ts[n + 1]);
      for (int r = 0; r < 8; ++r)
        dwm[r] = dW[((size_t)(n + 1) * BATCH + bbase + r + 8 * hi) * NZ + kcol] * sc;
    }
    float t1 = ts[n + 1];

    { // layer 1: h1 = lipswish(zhat @ W0 + gb0 + t1*gw0[0])
      float bias = gb0j + t1 * gw00j;
      v8f acc = {};
      for (int kk = 0; kk < 2; ++kk)
        acc = wmma_bf16(ld_frag(H2 + (kk * 32 + lane) * 16),
                        ld_frag(W0 + ((wv * 2 + kk) * 32 + lane) * 16), acc);
      for (int r = 0; r < 8; ++r)
        scatterA(H1, r + 8 * hi, wv * 16 + kcol, f2bfu(lipswish(acc[r] + bias)));
    }
    __syncthreads();
    { // layer 2
      v8f acc = {};
      for (int kb = 0; kb < 4; ++kb)
        acc = wmma_bf16(ld_frag(H1 + (kb * 32 + lane) * 16),
                        ld_frag(W1 + ((wv * 4 + kb) * 32 + lane) * 16), acc);
      for (int r = 0; r < 8; ++r)
        scatterA(H2, r + 8 * hi, wv * 16 + kcol, f2bfu(lipswish(acc[r] + gb1j)));
    }
    __syncthreads();
    { // layer 3: g = tanh(h2 @ W2 + gb2); fuse both dW contractions.
      // N-tile nt of the 1024-wide output is exactly hidden index h = nt,
      // lane%16 is the noise index -> reduce across the 16-lane half-groups.
      v16bf a0 = ld_frag(H2 + (0 * 32 + lane) * 16);
      v16bf a1 = ld_frag(H2 + (1 * 32 + lane) * 16);
      v16bf a2 = ld_frag(H2 + (2 * 32 + lane) * 16);
      v16bf a3 = ld_frag(H2 + (3 * 32 + lane) * 16);
      for (int ti = 0; ti < 8; ++ti) {
        int nt = wv + ti * 8;
        v8f acc = {};
        acc = wmma_bf16(a0, ld_frag(W2 + ((nt * 4 + 0) * 32 + lane) * 16), acc);
        acc = wmma_bf16(a1, ld_frag(W2 + ((nt * 4 + 1) * 32 + lane) * 16), acc);
        acc = wmma_bf16(a2, ld_frag(W2 + ((nt * 4 + 2) * 32 + lane) * 16), acc);
        acc = wmma_bf16(a3, ld_frag(W2 + ((nt * 4 + 3) * 32 + lane) * 16), acc);
        float gb = gb2r[ti];
        for (int r = 0; r < 8; ++r) {
          float g = fast_tanh(acc[r] + gb);
          float s1 = hasCur  ? g * dwn[r] : 0.f;
          float s2 = hasNext ? g * dwm[r] : 0.f;
          s1 += __shfl_xor(s1, 1); s1 += __shfl_xor(s1, 2);
          s1 += __shfl_xor(s1, 4); s1 += __shfl_xor(s1, 8);
          s2 += __shfl_xor(s2, 1); s2 += __shfl_xor(s2, 2);
          s2 += __shfl_xor(s2, 4); s2 += __shfl_xor(s2, 8);
          if (kcol == 0) {
            int b = r + 8 * hi;
            if (hasCur)  BUF1[b * 64 + nt] = s1;   // p_cur  = g.dWs_n
            if (hasNext) BUF0[b * 64 + nt] = s2;   // p_next = g.dWs_{n+1}
          }
        }
      }
    }
    __syncthreads();
    if (n >= 0) {
      const float4 pc = *(const float4*)&BUF1[tid * 4];
      z[0] += 0.5f * pc.x; z[1] += 0.5f * pc.y;
      z[2] += 0.5f * pc.z; z[3] += 0.5f * pc.w;
      readout(out, bbase + bown, n + 1, z, rwreg, rbv, tid);
    }
  }
}

extern "C" void kernel_launch(void* const* d_in, const int* in_sizes, int n_in,
                              void* d_out, int out_size, void* d_ws, size_t ws_size,
                              hipStream_t stream) {
  (void)in_sizes; (void)n_in; (void)out_size; (void)ws_size;
  const float* ts   = (const float*)d_in[0];
  const float* xin  = (const float*)d_in[2];
  const float* dW   = (const float*)d_in[3];
  const float* iw0  = (const float*)d_in[4];
  const float* ib0  = (const float*)d_in[5];
  const float* iw1  = (const float*)d_in[6];
  const float* ib1  = (const float*)d_in[7];
  const float* iw2  = (const float*)d_in[8];
  const float* ib2  = (const float*)d_in[9];
  const float* gw0  = (const float*)d_in[10];
  const float* gb0  = (const float*)d_in[11];
  const float* gw1  = (const float*)d_in[12];
  const float* gb1  = (const float*)d_in[13];
  const float* gw2  = (const float*)d_in[14];
  const float* gb2  = (const float*)d_in[15];
  const float* rw   = (const float*)d_in[16];
  const float* rb   = (const float*)d_in[17];
  float* out  = (float*)d_out;
  float* x0ws = (float*)d_ws;                   // 4096*64 f32 = 1 MB scratch

  const size_t shm1 = 37888u * 2u;              // 75776 B
  const size_t shm2 = 327680u;                  // 320 KB (CDNA5 WGP max)
  (void)hipFuncSetAttribute((const void*)sdegan_init_mlp,
                            hipFuncAttributeMaxDynamicSharedMemorySize, (int)shm1);
  (void)hipFuncSetAttribute((const void*)sdegan_scan,
                            hipFuncAttributeMaxDynamicSharedMemorySize, (int)shm2);

  sdegan_init_mlp<<<dim3(BATCH / 16), dim3(256), shm1, stream>>>(
      xin, iw0, ib0, iw1, ib1, iw2, ib2, x0ws);
  sdegan_scan<<<dim3(BATCH / 16), dim3(256), shm2, stream>>>(
      ts, dW, gw0, gb0, gw1, gb1, gw2, gb2, rw, rb, x0ws, out);
}
